// CrAKNAttention_91139206021792
// MI455X (gfx1250) — matrix operand order repaired and checked by reading, not compiled, not run on hardware
//
#include <hip/hip_runtime.h>
#include <hip/hip_bf16.h>

// ---------------- problem constants ----------------
constexpr int S_   = 768;   // seq len
constexpr int DIN_ = 128;   // input dim
constexpr int H_   = 8;     // heads
constexpr int HD_  = 128;   // H*DH
constexpr int NROWS_  = S_ * S_;        // 589824 flattened pair-rows
constexpr int NTILES_ = NROWS_ / 16;    // 36864 16-row tiles

// ---------------- vector types ----------------
typedef _Float16 half16_t __attribute__((ext_vector_type(16)));
typedef _Float16 half8_t  __attribute__((ext_vector_type(8)));
typedef float    f32x8_t  __attribute__((ext_vector_type(8)));
typedef float    f32x4_t  __attribute__((ext_vector_type(4)));

union H16u { half16_t v; half8_t h8[2]; _Float16 h[16]; };

__device__ __forceinline__ float mish_f(float x) {
    // mish(x) = x * tanh(softplus(x)); with e = exp(x):
    // tanh(ln(1+e)) = ((1+e)^2 - 1)/((1+e)^2 + 1) = 1 - 2/((1+e)^2 + 1)
    // Hardware v_rcp_f32 (~1 ulp) instead of IEEE divide: data path is
    // already f16-quantized, so the divide's extra precision is wasted VALU.
    float e = __expf(x);
    float w = 1.0f + e;                  // overflow -> inf -> rcp -> 0 -> t = 1
    float r = __builtin_amdgcn_rcpf(w * w + 1.0f);
    float t = 1.0f - 2.0f * r;
    return x * t;
}

__device__ __forceinline__ float dot4(f32x4_t a, f32x4_t b) {
    return a[0]*b[0] + a[1]*b[1] + a[2]*b[2] + a[3]*b[3];
}

// ================= kernel 1: qkv = x @ Wqkv^T + bqkv =================
__global__ void __launch_bounds__(256) crakn_qkv(
    const float* __restrict__ x, const float* __restrict__ Wqkv,
    const float* __restrict__ bqkv, float* __restrict__ qkv)
{
    int idx = blockIdx.x * 256 + threadIdx.x;      // S*384 outputs
    if (idx >= S_ * 384) return;
    int s = idx / 384, j = idx % 384;
    const float* xr = x    + (size_t)s * DIN_;
    const float* wr = Wqkv + (size_t)j * DIN_;
    float acc = bqkv[j];
    #pragma unroll 8
    for (int c = 0; c < DIN_; c += 4)
        acc += dot4(*(const f32x4_t*)(xr + c), *(const f32x4_t*)(wr + c));
    qkv[idx] = acc;
}

// ===== kernel 2 (fused): be = mish(bias@Wd^T+bd) -> norms + bias_out =====
// One wave handles one 16-row tile of the flattened [S*S, 128] matrix.
// be lives only in registers/LDS: single pass over the 302MB bias tensor.
// bias in / bias_out out are marked non-temporal: each is a one-shot
// >L2-sized stream, keep them from evicting weights + the diffs array.
__global__ void __launch_bounds__(128) crakn_bias_fused(
    const float* __restrict__ bias,
    const float* __restrict__ Wd, const float* __restrict__ bd,
    const float* __restrict__ Wb, const float* __restrict__ bb,
    float* __restrict__ bias_out,          // [S,S,DIN] (part of d_out)
    float* __restrict__ diffs)             // [H,S,S]   (workspace)
{
    extern __shared__ __align__(16) _Float16 smem[];
    _Float16* sWd = smem;                  // 128x128 f16 (row-major [n][k])
    _Float16* sWb = smem + 128 * 128;      // 128x128 f16
    _Float16* sBe = smem + 2 * 128 * 128;  // 4 waves x 16x128 f16

    const int tid = threadIdx.x;
    for (int i = tid; i < 128 * 128; i += 128) {
        sWd[i] = (_Float16)Wd[i];
        sWb[i] = (_Float16)Wb[i];
    }
    __syncthreads();

    const int lane = tid & 31;
    const int wave = tid >> 5;
    _Float16* sBeW = sBe + wave * (16 * 128);

    const int nl  = lane & 15;             // A row / B,C,D column within tile
    const int mb  = (lane < 16) ? 0 : 8;   // C/D row base (VGPR r -> row mb+r)
    const int kbA = (lane < 16) ? 0 : 8;   // 16-bit A-matrix K base
    const int kbB = (lane < 16) ? 0 : 8;   // 16-bit B-matrix K base

    float bdv[8], bbv[8];                  // per-lane bias entries, tile-invariant
    #pragma unroll
    for (int j = 0; j < 8; ++j) { bdv[j] = bd[j*16 + nl]; bbv[j] = bb[j*16 + nl]; }

    const int wavesTotal = gridDim.x * 4;
    for (int tile = blockIdx.x * 4 + wave; tile < NTILES_; tile += wavesTotal) {
        const size_t row0 = (size_t)tile * 16;
        const int s  = (int)(row0 / S_);   // 768 % 16 == 0 -> whole tile same s
        const int t0 = (int)(row0 % S_);

        // ---- A fragments: 16x128 bias tile, global f32 -> f16, A layout ----
        // Non-temporal: 302 MB one-shot stream, do not cache in L2.
        H16u afrag[4];
        const float* arow = bias + (row0 + (size_t)nl) * DIN_;
        #pragma unroll
        for (int kc = 0; kc < 4; ++kc) {
            const float* p = arow + kc * 32 + kbA;
            f32x4_t a0 = __builtin_nontemporal_load((const f32x4_t*)(p));
            f32x4_t a1 = __builtin_nontemporal_load((const f32x4_t*)(p + 4));
            f32x4_t a2 = __builtin_nontemporal_load((const f32x4_t*)(p + 16));
            f32x4_t a3 = __builtin_nontemporal_load((const f32x4_t*)(p + 20));
            #pragma unroll
            for (int i = 0; i < 4; ++i) {
                afrag[kc].h[i]      = (_Float16)a0[i];
                afrag[kc].h[4 + i]  = (_Float16)a1[i];
                afrag[kc].h[8 + i]  = (_Float16)a2[i];
                afrag[kc].h[12 + i] = (_Float16)a3[i];
            }
        }

        // ---- GEMM1: be tile = mish(bias @ Wd^T + bd) -> LDS (f16) ----
        #pragma unroll
        for (int j = 0; j < 8; ++j) {
            f32x8_t c = {0.f,0.f,0.f,0.f,0.f,0.f,0.f,0.f};
            #pragma unroll
            for (int kc = 0; kc < 4; ++kc) {
                const _Float16* pb = sWd + (size_t)(j*16 + nl) * 128 + kc * 32 + kbB;
                H16u bf;
                bf.h8[0] = *(const half8_t*)pb;
                bf.h8[1] = *(const half8_t*)(pb + 16);
                c = __builtin_amdgcn_wmma_f32_16x16x32_f16(
                        false, afrag[kc].v, false, bf.v, (short)0, c, false, false);
            }
            #pragma unroll
            for (int r = 0; r < 8; ++r) {
                float be = mish_f(c[r] + bdv[j]);
                sBeW[(mb + r) * 128 + (j * 16 + nl)] = (_Float16)be;
            }
        }

        // ---- per-(row,head) L2 norm -> diffs[h][s][t] (RT: reused by attn) ----
        #pragma unroll
        for (int i = 0; i < 4; ++i) {
            const int p  = lane * 4 + i;   // 0..127 = 16 rows x 8 heads
            const int m  = p >> 3;
            const int hh = p & 7;
            const _Float16* bp = sBeW + m * 128 + hh * 16;
            half8_t b0 = *(const half8_t*)bp;
            half8_t b1 = *(const half8_t*)(bp + 8);
            float ss = 0.f;
            #pragma unroll
            for (int d = 0; d < 8; ++d) {
                float f0 = (float)b0[d], f1 = (float)b1[d];
                ss += f0 * f0 + f1 * f1;
            }
            diffs[((size_t)hh * S_ + s) * S_ + (t0 + m)] = __builtin_amdgcn_sqrtf(ss);
        }

        // ---- A fragments for GEMM2 from the be tile in LDS ----
        H16u afrag2[4];
        #pragma unroll
        for (int kc = 0; kc < 4; ++kc) {
            const _Float16* pa = sBeW + nl * 128 + kc * 32 + kbA;
            afrag2[kc].h8[0] = *(const half8_t*)pa;
            afrag2[kc].h8[1] = *(const half8_t*)(pa + 16);
        }

        // ---- GEMM2: bias_out = mish(be @ Wb^T + bb) -> global (NT stores) ----
        float* orow = bias_out + row0 * DIN_;
        #pragma unroll
        for (int j = 0; j < 8; ++j) {
            f32x8_t c = {0.f,0.f,0.f,0.f,0.f,0.f,0.f,0.f};
            #pragma unroll
            for (int kc = 0; kc < 4; ++kc) {
                const _Float16* pb = sWb + (size_t)(j*16 + nl) * 128 + kc * 32 + kbB;
                H16u bf;
                bf.h8[0] = *(const half8_t*)pb;
                bf.h8[1] = *(const half8_t*)(pb + 16);
                c = __builtin_amdgcn_wmma_f32_16x16x32_f16(
                        false, afrag2[kc].v, false, bf.v, (short)0, c, false, false);
            }
            #pragma unroll
            for (int r = 0; r < 8; ++r) {
                float v = mish_f(c[r] + bbv[j]);
                __builtin_nontemporal_store(
                    v, orow + (size_t)(mb + r) * DIN_ + (j * 16 + nl));
            }
        }
    }
}

// ===== kernel 3: softmax attention; one block per (head, s) row =====
__global__ void __launch_bounds__(256) crakn_attn(
    const float* __restrict__ qkv, const float* __restrict__ diffs,
    float* __restrict__ vals)
{
    __shared__ float red[256];
    __shared__ float accs[16 * 256];
    const int tid = threadIdx.x;
    const int h = blockIdx.x / S_;
    const int s = blockIdx.x % S_;

    const float* qp = qkv + (size_t)s * 384 + h * 48;
    f32x4_t q0 = *(const f32x4_t*)(qp);
    f32x4_t q1 = *(const f32x4_t*)(qp + 4);
    f32x4_t q2 = *(const f32x4_t*)(qp + 8);
    f32x4_t q3 = *(const f32x4_t*)(qp + 12);

    const float* drow = diffs + ((size_t)h * S_ + s) * S_;
    float lg[3];
    float lmax = -3.4e38f;
    #pragma unroll
    for (int i = 0; i < 3; ++i) {
        int t = tid + i * 256;
        const float* kp = qkv + (size_t)t * 384 + h * 48 + 16;
        float dot = dot4(q0, *(const f32x4_t*)(kp))
                  + dot4(q1, *(const f32x4_t*)(kp + 4))
                  + dot4(q2, *(const f32x4_t*)(kp + 8))
                  + dot4(q3, *(const f32x4_t*)(kp + 12));
        lg[i] = 0.25f * dot + drow[t];     // 1/sqrt(16) = 0.25
        lmax = fmaxf(lmax, lg[i]);
    }
    red[tid] = lmax; __syncthreads();
    for (int st = 128; st > 0; st >>= 1) {
        if (tid < st) red[tid] = fmaxf(red[tid], red[tid + st]);
        __syncthreads();
    }
    const float m = red[0];
    __syncthreads();
    float e[3]; float lsum = 0.f;
    #pragma unroll
    for (int i = 0; i < 3; ++i) { e[i] = __expf(lg[i] - m); lsum += e[i]; }
    red[tid] = lsum; __syncthreads();
    for (int st = 128; st > 0; st >>= 1) {
        if (tid < st) red[tid] += red[tid + st];
        __syncthreads();
    }
    const float inv = 1.0f / red[0];

    float a[16];
    #pragma unroll
    for (int d = 0; d < 16; ++d) a[d] = 0.f;
    #pragma unroll
    for (int i = 0; i < 3; ++i) {
        int t = tid + i * 256;
        const float w = e[i] * inv;
        const float* vp = qkv + (size_t)t * 384 + h * 48 + 32;
        f32x4_t v0 = *(const f32x4_t*)(vp);
        f32x4_t v1 = *(const f32x4_t*)(vp + 4);
        f32x4_t v2 = *(const f32x4_t*)(vp + 8);
        f32x4_t v3 = *(const f32x4_t*)(vp + 12);
        #pragma unroll
        for (int d = 0; d < 4; ++d) {
            a[d]      += w * v0[d];
            a[4 + d]  += w * v1[d];
            a[8 + d]  += w * v2[d];
            a[12 + d] += w * v3[d];
        }
    }
    #pragma unroll
    for (int d = 0; d < 16; ++d) accs[d * 256 + tid] = a[d];
    __syncthreads();
    for (int st = 128; st > 0; st >>= 1) {
        if (tid < st) {
            #pragma unroll
            for (int d = 0; d < 16; ++d)
                accs[d * 256 + tid] += accs[d * 256 + tid + st];
        }
        __syncthreads();
    }
    if (tid < 16) vals[(size_t)s * HD_ + h * 16 + tid] = accs[tid * 256];
}

// ===== kernel 4: out = vals @ Wo^T + bo =====
__global__ void __launch_bounds__(128) crakn_outproj(
    const float* __restrict__ vals, const float* __restrict__ Wo,
    const float* __restrict__ bo, float* __restrict__ out)
{
    const int s = blockIdx.x, j = threadIdx.x;
    const float* vr = vals + (size_t)s * HD_;
    const float* wr = Wo + (size_t)j * HD_;
    float acc = bo[j];
    #pragma unroll 8
    for (int c = 0; c < HD_; c += 4)
        acc += dot4(*(const f32x4_t*)(vr + c), *(const f32x4_t*)(wr + c));
    out[(size_t)s * DIN_ + j] = acc;
}

// ================= launcher =================
extern "C" void kernel_launch(void* const* d_in, const int* in_sizes, int n_in,
                              void* d_out, int out_size, void* d_ws, size_t ws_size,
                              hipStream_t stream) {
    const float* x    = (const float*)d_in[0];
    const float* bias = (const float*)d_in[1];
    const float* Wqkv = (const float*)d_in[2];
    const float* bqkv = (const float*)d_in[3];
    const float* Wd   = (const float*)d_in[4];
    const float* bd   = (const float*)d_in[5];
    const float* Wo   = (const float*)d_in[6];
    const float* bo   = (const float*)d_in[7];
    const float* Wb   = (const float*)d_in[8];
    const float* bb   = (const float*)d_in[9];

    float* out      = (float*)d_out;            // [S, DIN]
    float* bias_out = out + (size_t)S_ * DIN_;  // [S, S, DIN]

    // workspace: qkv [S,384] | vals [S,128] | diffs [H,S,S]  (~20 MB)
    float* qkv   = (float*)d_ws;
    float* vals  = qkv + (size_t)S_ * 384;
    float* diffs = vals + (size_t)S_ * HD_;

    crakn_qkv<<<(S_ * 384) / 256, 256, 0, stream>>>(x, Wqkv, bqkv, qkv);

    const size_t shmem = (size_t)(2 * 128 * 128 + 4 * 16 * 128) * sizeof(_Float16); // 80 KB
    crakn_bias_fused<<<2304, 128, shmem, stream>>>(bias, Wd, bd, Wb, bb,
                                                   bias_out, diffs);

    crakn_attn<<<H_ * S_, 256, 0, stream>>>(qkv, diffs, vals);
    crakn_outproj<<<S_, 128, 0, stream>>>(vals, Wo, bo, out);
}
